// ManiPointSegment3_44169443672119
// MI455X (gfx1250) — matrix operand order repaired and checked by reading, not compiled, not run on hardware
//
#include <hip/hip_runtime.h>
#include <hip/hip_bf16.h>
#include <math.h>

typedef __attribute__((ext_vector_type(16))) _Float16 v16h;
typedef __attribute__((ext_vector_type(8)))  _Float16 v8h;
typedef __attribute__((ext_vector_type(8)))  float    v8f;

#define BNC 0.9999950000374997f   /* 1/sqrt(1+1e-5) */
#define BB 8                      /* batch */

struct WnP { const float* p[12]; };

/* ---- fragment loaders: K-contiguous rows, two aligned 16B loads per fragment ---- */
__device__ inline v16h ldA_frag(const _Float16* rowbase, int kt, int lh) {
  /* A 16x32 fp16 layout: lane half 0 needs K = kt+[0..7], kt+16+[0..7];
                          lane half 1 needs K = kt+8+[0..7], kt+24+[0..7] */
  const v8h* r = (const v8h*)rowbase;
  v8h lo = r[(kt >> 3) + lh];
  v8h hi = r[(kt >> 3) + 2 + lh];
  return __builtin_shufflevector(lo, hi, 0, 1, 2, 3, 4, 5, 6, 7,
                                 8, 9, 10, 11, 12, 13, 14, 15);
}
__device__ inline v16h ldB_frag(const _Float16* rowbase, int kt, int lh) {
  /* B 32x16 fp16 layout: lane half lh needs K = kt+8*lh+[0..7], kt+16+8*lh+[0..7] */
  const v8h* r = (const v8h*)rowbase;
  v8h lo = r[(kt >> 3) + lh];
  v8h hi = r[(kt >> 3) + 2 + lh];
  return __builtin_shufflevector(lo, hi, 0, 1, 2, 3, 4, 5, 6, 7,
                                 8, 9, 10, 11, 12, 13, 14, 15);
}

/* ---------------- transpose (B,3,N) -> (B,N,3) ---------------- */
__global__ __launch_bounds__(256) void k_transpose(const float* __restrict__ in,
                                                   float* __restrict__ out, int N) {
  int t = blockIdx.x * 256 + threadIdx.x;
  if (t >= BB * 3 * N) return;
  int n = t % N; int c = (t / N) % 3; int b = t / (3 * N);
  out[(b * N + n) * 3 + c] = in[t];
}

/* ---------------- inverse density ---------------- */
__global__ __launch_bounds__(128) void k_density(const float* __restrict__ xyz, int N,
                                                 float bw, float* __restrict__ invd) {
  int tilesPerB = N / 128;
  int b = blockIdx.x / tilesPerB;
  int tile = blockIdx.x % tilesPerB;
  int t = threadIdx.x;
  int n = tile * 128 + t;
  __shared__ float sx[128 * 3];
  float px = xyz[(b * N + n) * 3 + 0];
  float py = xyz[(b * N + n) * 3 + 1];
  float pz = xyz[(b * N + n) * 3 + 2];
  float inv2 = 1.0f / (2.0f * bw * bw);
  float sum = 0.f;
  for (int m0 = 0; m0 < N; m0 += 128) {
    sx[t * 3 + 0] = xyz[(b * N + m0 + t) * 3 + 0];
    sx[t * 3 + 1] = xyz[(b * N + m0 + t) * 3 + 1];
    sx[t * 3 + 2] = xyz[(b * N + m0 + t) * 3 + 2];
    __syncthreads();
    for (int j = 0; j < 128; ++j) {
      float dx = px - sx[j * 3 + 0];
      float dy = py - sx[j * 3 + 1];
      float dz = pz - sx[j * 3 + 2];
      sum += expf(-(dx * dx + dy * dy + dz * dz) * inv2);
    }
    __syncthreads();
  }
  invd[b * N + n] = (2.5f * bw) * (float)N / sum;
}

/* ---------------- farthest point sampling (one block per batch) ---------------- */
__global__ __launch_bounds__(256) void k_fps(const float* __restrict__ xyz, int N, int S,
                                             int* __restrict__ outi) {
  int b = blockIdx.x, t = threadIdx.x;
  __shared__ float sd[2048];
  __shared__ float rv[256];
  __shared__ int   ri[256];
  __shared__ int   sfar;
  for (int n = t; n < N; n += 256) sd[n] = 1e10f;
  if (t == 0) sfar = 0;
  __syncthreads();
  for (int it = 0; it < S; ++it) {
    int far = sfar;
    if (t == 0) outi[b * S + it] = far;
    float cx = xyz[(b * N + far) * 3 + 0];
    float cy = xyz[(b * N + far) * 3 + 1];
    float cz = xyz[(b * N + far) * 3 + 2];
    float bv = -1.f; int bi = 0;
    for (int n = t; n < N; n += 256) {
      float dx = xyz[(b * N + n) * 3 + 0] - cx;
      float dy = xyz[(b * N + n) * 3 + 1] - cy;
      float dz = xyz[(b * N + n) * 3 + 2] - cz;
      float d = dx * dx + dy * dy + dz * dz;
      float m = fminf(sd[n], d);
      sd[n] = m;
      if (m > bv) { bv = m; bi = n; }
    }
    rv[t] = bv; ri[t] = bi;
    __syncthreads();
    for (int s2 = 128; s2 > 0; s2 >>= 1) {
      if (t < s2) {
        if (rv[t + s2] > rv[t] || (rv[t + s2] == rv[t] && ri[t + s2] < ri[t])) {
          rv[t] = rv[t + s2]; ri[t] = ri[t + s2];
        }
      }
      __syncthreads();
    }
    if (t == 0) sfar = ri[0];
    __syncthreads();
  }
}

/* ---------------- gather sampled centroids ---------------- */
__global__ __launch_bounds__(256) void k_gather_xyz(const float* __restrict__ xyz,
                                                    const int* __restrict__ fi,
                                                    float* __restrict__ nx, int S, int N) {
  int t = blockIdx.x * 256 + threadIdx.x;
  if (t >= BB * S * 3) return;
  int j = t % 3; int s = (t / 3) % S; int b = t / (3 * S);
  nx[(b * S + s) * 3 + j] = xyz[(b * N + fi[b * S + s]) * 3 + j];
}

/* ---------------- k nearest neighbors (one thread per centroid) ---------------- */
__global__ __launch_bounds__(128) void k_knn(const float* __restrict__ xyz,
                                             const float* __restrict__ cen, int N, int S,
                                             int ns, int* __restrict__ outi) {
  int g = blockIdx.x * 128 + threadIdx.x;
  if (g >= BB * S) return;
  int b = g / S;
  float cx = cen[g * 3 + 0], cy = cen[g * 3 + 1], cz = cen[g * 3 + 2];
  float nd[64]; int ni[64];
  for (int i = 0; i < ns; ++i) { nd[i] = 3.0e38f; ni[i] = 0; }
  for (int m = 0; m < N; ++m) {
    float dx = xyz[(b * N + m) * 3 + 0] - cx;
    float dy = xyz[(b * N + m) * 3 + 1] - cy;
    float dz = xyz[(b * N + m) * 3 + 2] - cz;
    float d = dx * dx + dy * dy + dz * dz;
    if (d < nd[ns - 1]) {
      int pos = ns - 1;
      while (pos > 0 && nd[pos - 1] > d) {
        nd[pos] = nd[pos - 1]; ni[pos] = ni[pos - 1]; --pos;
      }
      nd[pos] = d; ni[pos] = m;
    }
  }
  for (int i = 0; i < ns; ++i) outi[g * ns + i] = ni[i];
}

/* ---------------- grouped coords + grouped inverse density ---------------- */
__global__ __launch_bounds__(256) void k_group_aux(const float* __restrict__ xyz,
                                                   const float* __restrict__ nx,
                                                   const int* __restrict__ idx,
                                                   const float* __restrict__ invd,
                                                   float* __restrict__ gxyz,
                                                   float* __restrict__ gdens,
                                                   int S, int K, int N, int gall) {
  int t = blockIdx.x * 256 + threadIdx.x;
  if (t >= BB * S * K) return;
  int kk = t % K; int s = (t / K) % S; int b = t / (K * S);
  int bs = b * S + s;
  int iv = gall ? kk : idx[bs * K + kk];
  float cx = 0.f, cy = 0.f, cz = 0.f;
  if (!gall) { cx = nx[bs * 3]; cy = nx[bs * 3 + 1]; cz = nx[bs * 3 + 2]; }
  gxyz[(bs * K + kk) * 3 + 0] = xyz[(b * N + iv) * 3 + 0] - cx;
  gxyz[(bs * K + kk) * 3 + 1] = xyz[(b * N + iv) * 3 + 1] - cy;
  gxyz[(bs * K + kk) * 3 + 2] = xyz[(b * N + iv) * 3 + 2] - cz;
  gdens[bs * K + kk] = invd[b * N + iv];
}

/* ---------------- per-group density max ---------------- */
__global__ __launch_bounds__(128) void k_gdmax(const float* __restrict__ gdens, int K,
                                               float* __restrict__ gdmax) {
  int bs = blockIdx.x, t = threadIdx.x;
  __shared__ float red[128];
  float v = -1e30f;
  for (int k = t; k < K; k += 128) v = fmaxf(v, gdens[bs * K + k]);
  red[t] = v;
  __syncthreads();
  for (int s2 = 64; s2 > 0; s2 >>= 1) {
    if (t < s2) red[t] = fmaxf(red[t], red[t + s2]);
    __syncthreads();
  }
  if (t == 0) gdmax[bs] = red[0];
}

/* ---- gather grouped features to fp16, K-padded rows (GEMM A operand) ---- */
__global__ __launch_bounds__(256) void k_featgather(const float* __restrict__ gxyz,
                                                    const float* __restrict__ pts,
                                                    const int* __restrict__ idx,
                                                    _Float16* __restrict__ feats,
                                                    int S, int K, int N, int Cp, int Kpad) {
  int t = blockIdx.x * 256 + threadIdx.x;
  int total = BB * S * K * Kpad;
  if (t >= total) return;
  int ci = t % Kpad; int r = t / Kpad;
  int kk = r % K; r /= K;
  int s = r % S; int b = r / S;
  int bs = b * S + s;
  float v = 0.f;
  if (ci < 3) v = gxyz[(bs * K + kk) * 3 + ci];
  else if (ci < Cp + 3) {
    int iv = idx ? idx[bs * K + kk] : kk;
    v = pts[(b * N + iv) * Cp + (ci - 3)];
  }
  feats[(bs * K + kk) * Kpad + ci] = (_Float16)v;
}

/* ---- mlp weight -> fp16, K-padded rows (GEMM B operand) ---- */
__global__ __launch_bounds__(256) void k_cvtpad(const float* __restrict__ W,
                                                _Float16* __restrict__ Wp,
                                                int C, int cin, int Kpad) {
  int t = blockIdx.x * 256 + threadIdx.x;
  if (t >= C * Kpad) return;
  int ci = t % Kpad; int co = t / Kpad;
  Wp[t] = (ci < cin) ? (_Float16)W[co * cin + ci] : (_Float16)0.0f;
}

/* ---- WMMA grouped-MLP GEMM: hT[bs][co][kk] = relu(BN(feats @ W^T)) * dens ratio ---- */
__global__ __launch_bounds__(256) void k_mlpw(const _Float16* __restrict__ feats,
                                              const _Float16* __restrict__ Wp,
                                              const float* __restrict__ mb,
                                              const float* __restrict__ mg,
                                              const float* __restrict__ mbe,
                                              const float* __restrict__ gdens,
                                              const float* __restrict__ gdmax,
                                              _Float16* __restrict__ hT,
                                              int M, int C, int K, int Kd) {
  int warp = threadIdx.x >> 5, lane = threadIdx.x & 31;
  int tilesM = M >> 4, tilesN = C >> 4;
  int gt = blockIdx.x * 8 + warp;
  if (gt >= tilesM * tilesN) return;
  int tm = gt % tilesM, tn = gt / tilesM;
  int m0 = tm << 4, n0 = tn << 4;
  int lh = lane >> 4, lm = lane & 15;
  const _Float16* rowA = feats + (size_t)(m0 + lm) * Kd;
  const _Float16* rowB = Wp + (size_t)(n0 + lm) * Kd;
  v8f acc = {};
  for (int kt = 0; kt < Kd; kt += 32) {
    v16h a = ldA_frag(rowA, kt, lh);
    v16h bm = ldB_frag(rowB, kt, lh);
    acc = __builtin_amdgcn_wmma_f32_16x16x32_f16(false, a, false, bm, (short)0, acc,
                                                 false, false);
  }
#pragma unroll
  for (int r = 0; r < 8; ++r) {
    int m = m0 + r + (lh ? 8 : 0);
    int n = n0 + lm;
    float v = acc[r] + mb[n];
    v = fmaxf(v * (mg[n] * BNC) + mbe[n], 0.f);
    int bs = m / K, kk = m % K;         /* gdens index == m */
    v *= gdens[m] / gdmax[bs];
    hT[((size_t)bs * C + n) * K + kk] = (_Float16)v;
  }
}

/* ---------------- weight network 3->8->8->16, fp16 out (transposed [bs][o][K]) -------- */
__global__ __launch_bounds__(128) void k_wn(const float* __restrict__ gxyz, WnP wp,
                                            _Float16* __restrict__ wT, int total, int K) {
  int g = blockIdx.x * 128 + threadIdx.x;
  if (g >= total) return;
  int bs = g / K, kk = g % K;
  float x0 = gxyz[g * 3 + 0], x1 = gxyz[g * 3 + 1], x2 = gxyz[g * 3 + 2];
  float a[8], tm[8];
  const float *W = wp.p[0], *bb = wp.p[1], *gg = wp.p[2], *be = wp.p[3];
  for (int o = 0; o < 8; ++o) {
    float v = W[o * 3] * x0 + W[o * 3 + 1] * x1 + W[o * 3 + 2] * x2 + bb[o];
    a[o] = fmaxf(v * (gg[o] * BNC) + be[o], 0.f);
  }
  W = wp.p[4]; bb = wp.p[5]; gg = wp.p[6]; be = wp.p[7];
  for (int o = 0; o < 8; ++o) {
    float v = bb[o];
    for (int c = 0; c < 8; ++c) v += W[o * 8 + c] * a[c];
    tm[o] = fmaxf(v * (gg[o] * BNC) + be[o], 0.f);
  }
  W = wp.p[8]; bb = wp.p[9]; gg = wp.p[10]; be = wp.p[11];
  for (int o = 0; o < 16; ++o) {
    float v = bb[o];
    for (int c = 0; c < 8; ++c) v += W[o * 8 + c] * tm[c];
    wT[((size_t)bs * 16 + o) * K + kk] = (_Float16)fmaxf(v * (gg[o] * BNC) + be[o], 0.f);
  }
}

/* -------- WMMA pair-einsum: X[bs, c*16+w] = sum_k hT[bs,c,k] * wT[bs,w,k] -------- */
__global__ __launch_bounds__(256) void k_pair(const _Float16* __restrict__ hT,
                                              const _Float16* __restrict__ wT,
                                              _Float16* __restrict__ X16,
                                              int S, int K, int C) {
  int warp = threadIdx.x >> 5, lane = threadIdx.x & 31;
  int ctN = C >> 4;
  int gt = blockIdx.x * 8 + warp;
  if (gt >= BB * S * ctN) return;
  int bs = gt / ctN;
  int c0 = (gt % ctN) << 4;
  int lh = lane >> 4, lm = lane & 15;
  const _Float16* rowA = hT + ((size_t)bs * C + c0 + lm) * K;   /* A[m][k]=h[k,c0+m] */
  const _Float16* rowB = wT + ((size_t)bs * 16 + lm) * K;       /* B[k][n]=w[k,n]    */
  v8f acc = {};
  for (int kt = 0; kt < K; kt += 32) {
    v16h a = ldA_frag(rowA, kt, lh);
    v16h bm = ldB_frag(rowB, kt, lh);
    acc = __builtin_amdgcn_wmma_f32_16x16x32_f16(false, a, false, bm, (short)0, acc,
                                                 false, false);
  }
#pragma unroll
  for (int r = 0; r < 8; ++r) {
    int m = r + (lh ? 8 : 0);
    X16[(size_t)bs * (C * 16) + (c0 + m) * 16 + lm] = (_Float16)acc[r];
  }
}

/* ---------------- fp32 -> fp16 convert ---------------- */
__global__ __launch_bounds__(256) void k_cvt16(const float* __restrict__ src,
                                               _Float16* __restrict__ dst, int n) {
  int t = blockIdx.x * 256 + threadIdx.x;
  if (t < n) dst[t] = (_Float16)src[t];
}

/* ---- WMMA lin GEMM: out = relu((X @ Wl^T + bl) * g*BN + be), fp32 out ---- */
__global__ __launch_bounds__(256) void k_lin(const _Float16* __restrict__ X,
                                             const _Float16* __restrict__ Wt,
                                             const float* __restrict__ bl,
                                             const float* __restrict__ g,
                                             const float* __restrict__ be,
                                             float* __restrict__ outp,
                                             int M, int Nc, int Kd) {
  int warp = threadIdx.x >> 5, lane = threadIdx.x & 31;
  int tilesM = (M + 15) >> 4, tilesN = Nc >> 4;
  int gt = blockIdx.x * 8 + warp;
  if (gt >= tilesM * tilesN) return;
  int tm = gt % tilesM, tn = gt / tilesM;
  int m0 = tm << 4, n0 = tn << 4;
  int lh = lane >> 4, lm = lane & 15;
  int arow = m0 + lm;
  bool aok = arow < M;
  const _Float16* rowA = X + (size_t)(aok ? arow : 0) * Kd;
  const _Float16* rowB = Wt + (size_t)(n0 + lm) * Kd;
  v8f acc = {};
  v16h zero = {};
  for (int kt = 0; kt < Kd; kt += 32) {
    if (aok) {
      int pf = kt + 128 < Kd ? kt + 128 : Kd - 1;
      __builtin_prefetch(&rowA[pf], 0, 1);   /* global_prefetch_b8 */
    }
    v16h a = aok ? ldA_frag(rowA, kt, lh) : zero;
    v16h bm = ldB_frag(rowB, kt, lh);
    acc = __builtin_amdgcn_wmma_f32_16x16x32_f16(false, a, false, bm, (short)0, acc,
                                                 false, false);
  }
#pragma unroll
  for (int r = 0; r < 8; ++r) {
    int m = m0 + r + (lh ? 8 : 0);
    int n = n0 + lm;
    if (m < M) {
      float v = acc[r] + bl[n];
      v = v * (g[n] * BNC) + be[n];
      outp[m * Nc + n] = fmaxf(v, 0.f);
    }
  }
}

/* ---------------- head: conv1 + groupnorm + relu + conv2 + log_softmax ---------------- */
__global__ __launch_bounds__(256) void k_head(const float* __restrict__ xfeat,
                                              const float* __restrict__ W1,
                                              const float* __restrict__ b1,
                                              const float* __restrict__ W2,
                                              const float* __restrict__ b2,
                                              const float* __restrict__ gg,
                                              const float* __restrict__ gb,
                                              float* __restrict__ out) {
  int b = blockIdx.x, t = threadIdx.x;
  __shared__ float xs[512];
  __shared__ float hn[256];
  __shared__ float red[256];
  xs[t] = xfeat[b * 256 + t];
  xs[256 + t] = xfeat[BB * 256 + b * 256 + t];
  __syncthreads();
  float acc = b1[t];
  for (int c = 0; c < 512; ++c) acc += W1[t * 512 + c] * xs[c];
  red[t] = acc;
  __syncthreads();
  for (int s2 = 128; s2 > 0; s2 >>= 1) { if (t < s2) red[t] += red[t + s2]; __syncthreads(); }
  float mu = red[0] / 256.f;
  __syncthreads();
  red[t] = acc * acc;
  __syncthreads();
  for (int s2 = 128; s2 > 0; s2 >>= 1) { if (t < s2) red[t] += red[t + s2]; __syncthreads(); }
  float var = red[0] / 256.f - mu * mu;
  __syncthreads();
  float v = (acc - mu) * rsqrtf(var + 1e-5f) * gg[t] + gb[t];
  hn[t] = fmaxf(v, 0.f);
  __syncthreads();
  red[t] = W2[t] * hn[t];
  __syncthreads();
  for (int s2 = 128; s2 > 0; s2 >>= 1) { if (t < s2) red[t] += red[t + s2]; __syncthreads(); }
  float h20 = red[0] + b2[0];
  __syncthreads();
  red[t] = W2[256 + t] * hn[t];
  __syncthreads();
  for (int s2 = 128; s2 > 0; s2 >>= 1) { if (t < s2) red[t] += red[t + s2]; __syncthreads(); }
  float h21 = red[0] + b2[1];
  float mx = fmaxf(h20, h21);
  float lse = mx + logf(expf(h20 - mx) + expf(h21 - mx));
  float l0 = h20 - lse, l1 = h21 - lse;
  for (int n = t; n < 1024; n += 256) {
    out[b * 2048 + n] = l0;
    out[b * 2048 + 1024 + n] = l1;
  }
}

/* ================= host side ================= */
struct SAPtrs {
  const float *bg, *bbe, *Wl, *bl, *mW, *mb, *mg, *mbe;
  WnP wn;
};

static inline unsigned cdiv(long a, long b) { return (unsigned)((a + b - 1) / b); }

static void run_sa(hipStream_t st, const float* xyz, const float* pts, int N, int Cp,
                   int S, int K, int C, float bw, bool gall, const SAPtrs& P,
                   float* out_xyz, float* out_pts, float* invd, int* fpsI, int* knnI,
                   float* gxyz, float* gdens, float* gdmax, _Float16* feats16,
                   _Float16* Wm16, _Float16* hT16, _Float16* wT16, _Float16* X16,
                   _Float16* Wl16) {
  k_density<<<dim3(BB * N / 128), dim3(128), 0, st>>>(xyz, N, bw, invd);
  if (!gall) {
    k_fps<<<dim3(BB), dim3(256), 0, st>>>(xyz, N, S, fpsI);
    k_gather_xyz<<<dim3(cdiv(BB * S * 3, 256)), dim3(256), 0, st>>>(xyz, fpsI, out_xyz, S, N);
    k_knn<<<dim3(cdiv(BB * S, 128)), dim3(128), 0, st>>>(xyz, out_xyz, N, S, K, knnI);
    k_group_aux<<<dim3(cdiv((long)BB * S * K, 256)), dim3(256), 0, st>>>(
        xyz, out_xyz, knnI, invd, gxyz, gdens, S, K, N, 0);
  } else {
    k_group_aux<<<dim3(cdiv((long)BB * S * K, 256)), dim3(256), 0, st>>>(
        xyz, nullptr, nullptr, invd, gxyz, gdens, S, K, N, 1);
  }
  k_gdmax<<<dim3(BB * S), dim3(128), 0, st>>>(gdens, K, gdmax);

  /* grouped MLP as WMMA GEMM */
  int cin = Cp + 3;
  int Kpad = ((cin + 31) / 32) * 32;
  k_cvtpad<<<dim3(cdiv((long)C * Kpad, 256)), dim3(256), 0, st>>>(P.mW, Wm16, C, cin, Kpad);
  k_featgather<<<dim3(cdiv((long)BB * S * K * Kpad, 256)), dim3(256), 0, st>>>(
      gxyz, pts, gall ? nullptr : knnI, feats16, S, K, N, Cp, Kpad);
  int Mg = BB * S * K;   /* always a multiple of 16 */
  long tg = (long)(Mg / 16) * (C / 16);
  k_mlpw<<<dim3(cdiv(tg, 8)), dim3(256), 0, st>>>(feats16, Wm16, P.mb, P.mg, P.mbe,
                                                  gdens, gdmax, hT16, Mg, C, K, Kpad);

  k_wn<<<dim3(cdiv(BB * S * K, 128)), dim3(128), 0, st>>>(gxyz, P.wn, wT16,
                                                          BB * S * K, K);
  k_pair<<<dim3(cdiv((long)BB * S * (C / 16), 8)), dim3(256), 0, st>>>(hT16, wT16, X16,
                                                                       S, K, C);
  int Kd = 16 * C;
  k_cvt16<<<dim3(cdiv((long)C * Kd, 256)), dim3(256), 0, st>>>(P.Wl, Wl16, C * Kd);
  int M = BB * S;
  long tiles = (long)((M + 15) / 16) * (C / 16);
  k_lin<<<dim3(cdiv(tiles, 8)), dim3(256), 0, st>>>(X16, Wl16, P.bl, P.bg, P.bbe,
                                                    out_pts, M, C, Kd);
}

extern "C" void kernel_launch(void* const* d_in, const int* in_sizes, int n_in,
                              void* d_out, int out_size, void* d_ws, size_t ws_size,
                              hipStream_t stream) {
  (void)in_sizes; (void)n_in; (void)out_size; (void)ws_size;
  auto F = [&](int i) { return (const float*)d_in[i]; };
  const float* cloud[2] = {F(0), F(1)};
  const float *hc1W = F(2), *hc1b = F(3), *hc2W = F(4), *hc2b = F(5), *gng = F(6), *gnb = F(7);
  auto mkSA = [&](int base) {
    SAPtrs p;
    p.bg = F(base); p.bbe = F(base + 1); p.Wl = F(base + 2); p.bl = F(base + 3);
    p.mW = F(base + 4); p.mb = F(base + 5); p.mg = F(base + 6); p.mbe = F(base + 7);
    for (int i = 0; i < 12; ++i) p.wn.p[i] = F(base + 8 + i);
    return p;
  };
  SAPtrs sa1 = mkSA(8), sa2 = mkSA(28), sa3 = mkSA(48);

  /* workspace bump allocator (256B aligned) */
  size_t off = 0;
  char* w = (char*)d_ws;
  auto A = [&](size_t bytes) {
    size_t o = off;
    off += (bytes + 255) & ~(size_t)255;
    return o;
  };
  float*    xyzT    = (float*)(w + A((size_t)BB * 2048 * 3 * 4));
  float*    invd    = (float*)(w + A((size_t)BB * 2048 * 4));
  int*      fpsI    = (int*)  (w + A((size_t)BB * 512 * 4));
  int*      knnI    = (int*)  (w + A((size_t)BB * 512 * 64 * 4));
  float*    gxyz    = (float*)(w + A((size_t)BB * 16384 * 3 * 4));
  float*    gdens   = (float*)(w + A((size_t)BB * 16384 * 4));
  float*    gdmax   = (float*)(w + A((size_t)BB * 512 * 4));
  _Float16* feats16 = (_Float16*)(w + A((size_t)BB * 8192 * 96 * 2)); /* max S*K*Kpad */
  _Float16* Wm16    = (_Float16*)(w + A((size_t)256 * 160 * 2));
  _Float16* hT16    = (_Float16*)(w + A((size_t)BB * 1048576 * 2));
  _Float16* wT16    = (_Float16*)(w + A((size_t)BB * 16384 * 16 * 2));
  _Float16* X16     = (_Float16*)(w + A((size_t)BB * 524288 * 2));
  _Float16* Wl16    = (_Float16*)(w + A((size_t)1048576 * 2));
  float*    ptsA    = (float*)(w + A((size_t)BB * 512 * 64 * 4));
  float*    ptsB    = (float*)(w + A((size_t)BB * 128 * 128 * 4));
  float*    xyzA    = (float*)(w + A((size_t)BB * 512 * 3 * 4));
  float*    xyzB    = (float*)(w + A((size_t)BB * 128 * 3 * 4));
  float*    xfeat   = (float*)(w + A((size_t)2 * BB * 256 * 4));

  for (int br = 0; br < 2; ++br) {
    k_transpose<<<dim3(cdiv(BB * 3 * 2048, 256)), dim3(256), 0, stream>>>(cloud[br], xyzT, 2048);
    /* SA1: N=2048, Cpts=3 (points=xyz), S=512, K=32, C=64, bw=0.1, Kpad=32 */
    run_sa(stream, xyzT, xyzT, 2048, 3, 512, 32, 64, 0.1f, false, sa1, xyzA, ptsA,
           invd, fpsI, knnI, gxyz, gdens, gdmax, feats16, Wm16, hT16, wT16, X16, Wl16);
    /* SA2: N=512, Cpts=64, S=128, K=64, C=128, bw=0.2, Kpad=96 */
    run_sa(stream, xyzA, ptsA, 512, 64, 128, 64, 128, 0.2f, false, sa2, xyzB, ptsB,
           invd, fpsI, knnI, gxyz, gdens, gdmax, feats16, Wm16, hT16, wT16, X16, Wl16);
    /* SA3 (group_all): N=K=128, Cpts=128, S=1, C=256, bw=0.4, Kpad=160 */
    run_sa(stream, xyzB, ptsB, 128, 128, 1, 128, 256, 0.4f, true, sa3, nullptr,
           xfeat + br * BB * 256, invd, fpsI, knnI, gxyz, gdens, gdmax, feats16, Wm16,
           hT16, wT16, X16, Wl16);
  }
  k_head<<<dim3(BB), dim3(256), 0, stream>>>(xfeat, hc1W, hc1b, hc2W, hc2b, gng, gnb,
                                             (float*)d_out);
}
